// LifecycleStateUpdater_90022514524503
// MI455X (gfx1250) — matrix operand order repaired and checked by reading, not compiled, not run on hardware
//
#include <hip/hip_runtime.h>
#include <cstdint>
#include <cstddef>

// ---------------------------------------------------------------------------
// MI455X (gfx1250, wave32) implementation of LifecycleStateUpdater.
//   proj   = relu(event_X[lc_evt_idx] @ W_proj.T + b_proj)      (bf16 WMMA)
//   profile= scatter_mean(proj, lc_obj_idx)                      (f32 atomics)
//   GRUCell(profile, object_X) -> masked blend by main_idx       (bf16 WMMA)
// All GEMMs use v_wmma_f32_16x16x32_bf16 (f32 accumulate).
// ---------------------------------------------------------------------------

typedef __attribute__((ext_vector_type(16))) __bf16 v16bf;
typedef __attribute__((ext_vector_type(8)))  float  v8f;

#define HDIM 256            // obj_dim == evt_dim == hidden
#define KCH  8              // 256 / 32 K-chunks
#define LDSS 264            // padded LDS row stride (ushorts): 264*2B -> bank-conflict free

__device__ __forceinline__ unsigned short f2bf(float f) {
  unsigned int u = __float_as_uint(f);
  unsigned int r = (u + 0x7FFFu + ((u >> 16) & 1u)) >> 16;   // RNE
  return (unsigned short)r;
}

union FragBF { unsigned int u[8]; v16bf v; };
union AccF   { float f[8];        v8f   v; };

__device__ __forceinline__ v8f wmma_bf16(v16bf a, v16bf b, v8f c) {
  // (neg_a, A, neg_b, B, c_mod, C, reuse_a, reuse_b)
  return __builtin_amdgcn_wmma_f32_16x16x32_bf16(false, a, false, b, (short)0, c, false, false);
}

// A-matrix 16x32 bf16 fragment from LDS (ISA 7.12.2 layout):
// lanes 0-15 (kh=0): VGPR0..3 -> K=0..7 pairs, VGPR4..7 -> K=16..23
// lanes 16-31 (kh=1): +8
__device__ __forceinline__ v16bf load_a_frag(const unsigned short* lds, int r, int kb, int kh) {
  FragBF a;
#pragma unroll
  for (int j = 0; j < 8; ++j) {
    int ko = ((j < 4) ? (2 * j) : (16 + 2 * (j - 4))) + 8 * kh;
    a.u[j] = *(const unsigned int*)(lds + r * LDSS + kb + ko);
  }
  return a.v;
}

// B-matrix 32x16 bf16 fragment from the pre-packed weight array:
// frag layout = [fragIdx][lane][16 bf16] -> each lane reads 32 contiguous bytes.
__device__ __forceinline__ v16bf load_b_frag(const unsigned short* wb, int fragIdx, int lane) {
  FragBF b;
  const uint4* p = (const uint4*)(wb + ((size_t)fragIdx * 32 + lane) * 16);
  uint4 lo = p[0];
  uint4 hi = p[1];
  b.u[0] = lo.x; b.u[1] = lo.y; b.u[2] = lo.z; b.u[3] = lo.w;
  b.u[4] = hi.x; b.u[5] = hi.y; b.u[6] = hi.z; b.u[7] = hi.w;
  return b.v;
}

__device__ __forceinline__ void atomic_add_f32(float* p, float v) {
  unsafeAtomicAdd(p, v);   // global_atomic_add_f32 (no return)
}

__device__ __forceinline__ float sigmoidf_(float x) { return 1.0f / (1.0f + __expf(-x)); }

// --------------------------- utility kernels -------------------------------

__global__ void zero_kernel(float* p, size_t n) {
  size_t i = (size_t)blockIdx.x * blockDim.x + threadIdx.x;
  size_t stride = (size_t)gridDim.x * blockDim.x;
  for (; i < n; i += stride) p[i] = 0.0f;
}

__global__ void mask_kernel(const int* mainIdx, float* mask, int nMain) {
  int i = blockIdx.x * blockDim.x + threadIdx.x;
  if (i < nMain) mask[mainIdx[i]] = 1.0f;
}

// Repack W [N][K] (f32, row-major, used as B = W.T so B[k][n] = W[n][k])
// into bf16 WMMA B-fragments: fragIdx = t*(K/32)+c, per-lane 16 contiguous bf16.
// B layout (32x16, 16-bit): lanes 0-15 hold K=0..7,16..23 for column n=lane,
// lanes 16-31 hold K=8..15,24..31 for column n=lane-16; pair j -> K = (j<4 ? 2j : 16+2(j-4)) (+8 hi).
__global__ void pack_weight_kernel(const float* __restrict__ W, unsigned short* __restrict__ out,
                                   int N, int K) {
  int kc = K / 32;
  int total = (N / 16) * kc * 32;
  int tid = blockIdx.x * blockDim.x + threadIdx.x;
  if (tid >= total) return;
  int lane = tid & 31;
  int c = (tid >> 5) % kc;
  int t = tid / (32 * kc);
  int n = t * 16 + (lane & 15);
  int kh = lane >> 4;
  unsigned short* o = out + ((size_t)(t * kc + c) * 32 + lane) * 16;
#pragma unroll
  for (int j = 0; j < 8; ++j) {
    int k = c * 32 + ((j < 4) ? (2 * j) : (16 + 2 * (j - 4))) + 8 * kh;
    o[2 * j + 0] = f2bf(W[(size_t)n * K + k + 0]);
    o[2 * j + 1] = f2bf(W[(size_t)n * K + k + 1]);
  }
}

// --------------------- edge projection + scatter-add -----------------------
// One block = 16 edges x 256 output dims. 4 waves, each wave owns 4 N-tiles.
__global__ __launch_bounds__(128) void edge_proj_scatter_kernel(
    const float* __restrict__ eventX, const int* __restrict__ lcEvt,
    const int* __restrict__ lcObj, const unsigned short* __restrict__ wbP,
    const float* __restrict__ bproj, float* __restrict__ sums,
    float* __restrict__ cnt, int nEdge) {
  __shared__ unsigned short evA[16 * LDSS];
  __shared__ int sEvt[16];
  __shared__ int sObj[16];

  const int e0 = blockIdx.x * 16;
  const int tid = threadIdx.x;

  if (tid < 16) sObj[tid] = (e0 + tid < nEdge) ? lcObj[e0 + tid] : -1;
  else if (tid < 32) { int r = tid - 16; sEvt[r] = (e0 + r < nEdge) ? lcEvt[e0 + r] : 0; }
  __syncthreads();

  // Gather 16 event rows, convert to bf16 in LDS (coalesced along k).
  for (int i = tid; i < 16 * HDIM; i += 128) {
    int r = i >> 8, k = i & (HDIM - 1);
    float v = eventX[(size_t)sEvt[r] * HDIM + k];
    evA[r * LDSS + k] = f2bf(v);
  }
  __syncthreads();

  const int wave = tid >> 5;
  const int lane = tid & 31;
  const int r = lane & 15;
  const int kh = lane >> 4;

  v8f zero = {};
  AccF acc[4];
#pragma unroll
  for (int j = 0; j < 4; ++j) acc[j].v = zero;

  for (int c = 0; c < KCH; ++c) {
    v16bf a = load_a_frag(evA, r, c * 32, kh);
#pragma unroll
    for (int j = 0; j < 4; ++j) {
      int t = wave * 4 + j;
      v16bf b = load_b_frag(wbP, t * KCH + c, lane);
      acc[j].v = wmma_bf16(a, b, acc[j].v);
    }
  }

  // Epilogue: bias + ReLU + atomic scatter into sums.
  // C layout: lane -> n = tile*16 + (lane&15); VGPR v -> m = (lane>>4)*8 + v.
#pragma unroll
  for (int j = 0; j < 4; ++j) {
    int n = (wave * 4 + j) * 16 + (lane & 15);
    float bias = bproj[n];
#pragma unroll
    for (int v = 0; v < 8; ++v) {
      int m = kh * 8 + v;
      int o = sObj[m];
      if (o >= 0) {
        float val = fmaxf(acc[j].f[v] + bias, 0.0f);
        atomic_add_f32(&sums[(size_t)o * HDIM + n], val);
      }
    }
  }
  if (tid < 16 && sObj[tid] >= 0) atomic_add_f32(&cnt[sObj[tid]], 1.0f);
}

// ----------------------------- GRU + blend ---------------------------------
// One block = 16 objects. 8 waves; wave w owns gate-tile triplet
// {2w,2w+1} (+0 / +16 / +32) of the 48 N-tiles (N=768), so the r/z/n gate
// values for output dim n live in the same lane/vgpr slot of this wave.
__global__ __launch_bounds__(256) void gru_blend_kernel(
    const float* __restrict__ objectX, const unsigned short* __restrict__ wbI,
    const unsigned short* __restrict__ wbH, const float* __restrict__ bih,
    const float* __restrict__ bhh, const float* __restrict__ sums,
    const float* __restrict__ cnt, const float* __restrict__ mask,
    float* __restrict__ out, int nObj) {
  __shared__ unsigned short profA[16 * LDSS];
  __shared__ unsigned short objA[16 * LDSS];

  const int o0 = blockIdx.x * 16;
  const int tid = threadIdx.x;

  // Build profile tile (sums / max(cnt,1)) and object tile in bf16 LDS.
  for (int i = tid; i < 16 * HDIM; i += 256) {
    int r = i >> 8, k = i & (HDIM - 1);
    int obj = o0 + r;
    float pv = 0.0f, hv = 0.0f;
    if (obj < nObj) {
      float inv = 1.0f / fmaxf(cnt[obj], 1.0f);
      pv = sums[(size_t)obj * HDIM + k] * inv;
      hv = objectX[(size_t)obj * HDIM + k];
    }
    profA[r * LDSS + k] = f2bf(pv);
    objA[r * LDSS + k]  = f2bf(hv);
  }
  __syncthreads();

  const int wave = tid >> 5;
  const int lane = tid & 31;
  const int r = lane & 15;
  const int kh = lane >> 4;

  v8f zero = {};
  AccF gi[6], gh[6];
#pragma unroll
  for (int i = 0; i < 6; ++i) { gi[i].v = zero; gh[i].v = zero; }

  for (int c = 0; c < KCH; ++c) {
    v16bf ap = load_a_frag(profA, r, c * 32, kh);
    v16bf ah = load_a_frag(objA,  r, c * 32, kh);
#pragma unroll
    for (int g = 0; g < 3; ++g) {
#pragma unroll
      for (int j = 0; j < 2; ++j) {
        int t = g * 16 + wave * 2 + j;           // N-tile within [0,48)
        int slot = g * 2 + j;
        v16bf bi = load_b_frag(wbI, t * KCH + c, lane);
        gi[slot].v = wmma_bf16(ap, bi, gi[slot].v);
        v16bf bh = load_b_frag(wbH, t * KCH + c, lane);
        gh[slot].v = wmma_bf16(ah, bh, gh[slot].v);
      }
    }
  }

  // Fused gates + masked blend. Output dims covered: n = (wave*2+j)*16+lane%16.
#pragma unroll
  for (int j = 0; j < 2; ++j) {
    int n = (wave * 2 + j) * 16 + (lane & 15);
    float bi_r = bih[n],        bh_r = bhh[n];
    float bi_z = bih[HDIM + n], bh_z = bhh[HDIM + n];
    float bi_n = bih[2 * HDIM + n], bh_n = bhh[2 * HDIM + n];
#pragma unroll
    for (int v = 0; v < 8; ++v) {
      int m = kh * 8 + v;
      int obj = o0 + m;
      if (obj < nObj) {
        float h  = objectX[(size_t)obj * HDIM + n];
        float rg = sigmoidf_(gi[j].f[v]     + bi_r + gh[j].f[v]     + bh_r);
        float zg = sigmoidf_(gi[2 + j].f[v] + bi_z + gh[2 + j].f[v] + bh_z);
        float ng = tanhf(gi[4 + j].f[v] + bi_n + rg * (gh[4 + j].f[v] + bh_n));
        float upd = (1.0f - zg) * ng + zg * h;
        float mk = mask[obj];
        out[(size_t)obj * HDIM + n] = mk * upd + (1.0f - mk) * h;
      }
    }
  }
}

// ------------------------------- launcher ----------------------------------

static inline size_t align_up(size_t x, size_t a) { return (x + a - 1) & ~(a - 1); }

extern "C" void kernel_launch(void* const* d_in, const int* in_sizes, int n_in,
                              void* d_out, int out_size, void* d_ws, size_t ws_size,
                              hipStream_t stream) {
  const float* objectX = (const float*)d_in[0];
  const float* eventX  = (const float*)d_in[1];
  const int*   lcObj   = (const int*)d_in[2];
  const int*   lcEvt   = (const int*)d_in[3];
  const int*   mainIdx = (const int*)d_in[4];
  const float* Wproj   = (const float*)d_in[5];
  const float* bproj   = (const float*)d_in[6];
  const float* Wih     = (const float*)d_in[7];
  const float* bih     = (const float*)d_in[8];
  const float* Whh     = (const float*)d_in[9];
  const float* bhh     = (const float*)d_in[10];
  float* out = (float*)d_out;

  const int nObj  = in_sizes[0] / HDIM;
  const int nEdge = in_sizes[2];
  const int nMain = in_sizes[4];

  // Workspace layout: sums | cnt | mask | bf16-packed W_proj, W_ih, W_hh
  char* ws = (char*)d_ws;
  size_t off = 0;
  float* sums = (float*)(ws + off); off += (size_t)nObj * HDIM * sizeof(float);
  float* cnt  = (float*)(ws + off); off += (size_t)nObj * sizeof(float);
  float* mask = (float*)(ws + off); off += (size_t)nObj * sizeof(float);
  off = align_up(off, 256);
  unsigned short* wbP = (unsigned short*)(ws + off); off += (size_t)HDIM * HDIM * sizeof(unsigned short);
  unsigned short* wbI = (unsigned short*)(ws + off); off += (size_t)3 * HDIM * HDIM * sizeof(unsigned short);
  unsigned short* wbH = (unsigned short*)(ws + off); off += (size_t)3 * HDIM * HDIM * sizeof(unsigned short);
  (void)ws_size; (void)n_in; (void)out_size;

  // 1) zero sums + cnt + mask (contiguous region).
  size_t nz = (size_t)nObj * (HDIM + 2);
  zero_kernel<<<2048, 256, 0, stream>>>(sums, nz);

  // 2) mask[main_idx] = 1
  mask_kernel<<<(nMain + 255) / 256, 256, 0, stream>>>(mainIdx, mask, nMain);

  // 3) repack weights to bf16 WMMA B-fragment order.
  {
    int tot = (HDIM / 16) * (HDIM / 32) * 32;          // W_proj  [256,256]
    pack_weight_kernel<<<(tot + 255) / 256, 256, 0, stream>>>(Wproj, wbP, HDIM, HDIM);
    int tot3 = (3 * HDIM / 16) * (HDIM / 32) * 32;     // W_ih/W_hh [768,256]
    pack_weight_kernel<<<(tot3 + 255) / 256, 256, 0, stream>>>(Wih, wbI, 3 * HDIM, HDIM);
    pack_weight_kernel<<<(tot3 + 255) / 256, 256, 0, stream>>>(Whh, wbH, 3 * HDIM, HDIM);
  }

  // 4) edge projection GEMM + scatter-add (dominant: 65.5 GFLOP on WMMA).
  int edgeBlocks = (nEdge + 15) / 16;
  edge_proj_scatter_kernel<<<edgeBlocks, 128, 0, stream>>>(
      eventX, lcEvt, lcObj, wbP, bproj, sums, cnt, nEdge);

  // 5) GRU gates (39.3 GFLOP on WMMA) + masked blend.
  int gruBlocks = (nObj + 15) / 16;
  gru_blend_kernel<<<gruBlocks, 256, 0, stream>>>(
      objectX, wbI, wbH, bih, bhh, sums, cnt, mask, out, nObj);
}
